// SelfAttention_31894427140594
// MI455X (gfx1250) — matrix operand order repaired and checked by reading, not compiled, not run on hardware
//
#include <hip/hip_runtime.h>
#include <hip/hip_fp16.h>

// B=2,T=8,L=512,C=1024,H=16,D=64. Tokens N=8192.
typedef _Float16 v16h __attribute__((ext_vector_type(16)));
typedef _Float16 v8h  __attribute__((ext_vector_type(8)));
typedef _Float16 v4h  __attribute__((ext_vector_type(4)));
typedef float    v8f  __attribute__((ext_vector_type(8)));

#define WMMA(a, b, c) \
  __builtin_amdgcn_wmma_f32_16x16x32_f16(false, (a), false, (b), (short)0, (c), false, false)

__device__ __forceinline__ v16h ld_frag(const _Float16* p0, const _Float16* p1) {
  v8h lo = *(const v8h*)p0;
  v8h hi = *(const v8h*)p1;
  v16h r;
#pragma unroll
  for (int i = 0; i < 8; ++i) { r[i] = lo[i]; r[i + 8] = hi[i]; }
  return r;
}

// ---------------------------------------------------------------------------
// Kernel 1: y[8192 x 3072] = f16( x @ [wq;wkv]^T + [bq;bkv] )
// block tile 128(M tokens) x 128(N outcols), K-step 32, 8 waves (2M x 4N),
// each wave: 4x2 tiles of v_wmma_f32_16x16x32_f16.
// ---------------------------------------------------------------------------
__global__ __launch_bounds__(256)
void qkv_gemm(const float* __restrict__ x, const float* __restrict__ wq,
              const float* __restrict__ bq, const float* __restrict__ wkv,
              const float* __restrict__ bkv, _Float16* __restrict__ y) {
  __shared__ _Float16 Xs[128 * 40];   // [m][k], pad 40 halves (80B rows, 16B aligned)
  __shared__ _Float16 Ws[128 * 40];   // [n][k]
  const int n0 = blockIdx.x * 128, m0 = blockIdx.y * 128;
  const int tid = threadIdx.x, wid = tid >> 5, lane = tid & 31;
  const int lnl = lane & 15, hf = lane >> 4;
  const int wm = wid >> 2, wn = wid & 3;
  const float* Wsrc = (n0 < 1024) ? (wq + (size_t)n0 * 1024)
                                  : (wkv + (size_t)(n0 - 1024) * 1024);
  const float* bsrc = (n0 < 1024) ? (bq + n0) : (bkv + (n0 - 1024));

  v8f acc[4][2];
#pragma unroll
  for (int a = 0; a < 4; a++)
#pragma unroll
    for (int b = 0; b < 2; b++)
#pragma unroll
      for (int i = 0; i < 8; i++) acc[a][b][i] = 0.f;

  for (int k0 = 0; k0 < 1024; k0 += 32) {
#pragma unroll
    for (int i = 0; i < 4; i++) {
      int idx = tid + i * 256;            // 1024 float4-slots: 128 rows x 8
      int row = idx >> 3, c4 = (idx & 7) * 4;
      float4 f = *(const float4*)(x + (size_t)(m0 + row) * 1024 + k0 + c4);
      v4h h = {(_Float16)f.x, (_Float16)f.y, (_Float16)f.z, (_Float16)f.w};
      *(v4h*)&Xs[row * 40 + c4] = h;
      float4 g = *(const float4*)(Wsrc + (size_t)row * 1024 + k0 + c4);
      v4h h2 = {(_Float16)g.x, (_Float16)g.y, (_Float16)g.z, (_Float16)g.w};
      *(v4h*)&Ws[row * 40 + c4] = h2;
    }
    __syncthreads();
    v16h af[4], bf[2];
#pragma unroll
    for (int mt = 0; mt < 4; mt++) {
      const _Float16* p = Xs + (wm * 64 + mt * 16 + lnl) * 40 + (hf ? 8 : 0);
      af[mt] = ld_frag(p, p + 16);        // A: K {0..7,16..23} / {8..15,24..31}
    }
#pragma unroll
    for (int nt = 0; nt < 2; nt++) {
      const _Float16* p = Ws + (wn * 32 + nt * 16 + lnl) * 40 + (hf ? 8 : 0);
      bf[nt] = ld_frag(p, p + 16);        // B: K {0..7,16..23} / {8..15,24..31}
    }
#pragma unroll
    for (int mt = 0; mt < 4; mt++)
#pragma unroll
      for (int nt = 0; nt < 2; nt++)
        acc[mt][nt] = WMMA(af[mt], bf[nt], acc[mt][nt]);
    __syncthreads();
  }

#pragma unroll
  for (int mt = 0; mt < 4; mt++) {
    int rbase = m0 + wm * 64 + mt * 16 + (hf ? 8 : 0);
#pragma unroll
    for (int nt = 0; nt < 2; nt++) {
      int cl = wn * 32 + nt * 16 + lnl;
      float bv = bsrc[cl];
      int col = n0 + cl;
#pragma unroll
      for (int i = 0; i < 8; i++)
        y[(size_t)(rbase + i) * 3072 + col] = (_Float16)(acc[mt][nt][i] + bv);
    }
  }
}

// ---------------------------------------------------------------------------
// Kernel 2: per (head, 128-query tile): l2norm q/k/v, S=q@k^T, online softmax,
// O += P@V. 8 waves x 16 query rows. 1/sqrt(D) folded into q scale.
// ---------------------------------------------------------------------------
__global__ __launch_bounds__(256)
void attn(const _Float16* __restrict__ y, _Float16* __restrict__ o) {
  __shared__ _Float16 Qs[128 * 72];       // [qrow][d]
  __shared__ _Float16 Ks[128 * 72];       // [krow][d]
  __shared__ _Float16 Vt[64 * 136];       // [d][krow]  (transposed for B-frags)
  __shared__ _Float16 Ps[8 * 16 * 136];   // per-wave P staging [row][kcol]

  const int bx = blockIdx.x;
  const int hid = bx >> 2, qt = bx & 3;
  const int bt = hid >> 4, h = hid & 15;
  const size_t tokbase = (size_t)bt * 512;
  const int tid = threadIdx.x, wid = tid >> 5, lane = tid & 31;
  const int lnl = lane & 15, hf = lane >> 4;

  // ---- load + normalize q tile (one thread per row) ----
  if (tid < 128) {
    const _Float16* src = y + (tokbase + qt * 128 + tid) * 3072 + h * 64;
    v8h vb[8];
    float ss = 0.f;
#pragma unroll
    for (int i = 0; i < 8; i++) {
      vb[i] = *(const v8h*)(src + i * 8);
#pragma unroll
      for (int j = 0; j < 8; j++) { float f = (float)vb[i][j]; ss += f * f; }
    }
    float sc = 0.125f / fmaxf(sqrtf(ss), 1e-12f);   // 1/sqrt(D) folded in
#pragma unroll
    for (int i = 0; i < 8; i++) {
      v8h sv;
#pragma unroll
      for (int j = 0; j < 8; j++) sv[j] = (_Float16)((float)vb[i][j] * sc);
      *(v8h*)&Qs[tid * 72 + i * 8] = sv;
    }
  }
  __syncthreads();

  // persistent q A-fragments (K = d: 0..31, 32..63)
  const int arow = wid * 16 + lnl;
  const int koffA = hf ? 8 : 0;
  v16h aq0 = ld_frag(Qs + arow * 72 + koffA,      Qs + arow * 72 + koffA + 16);
  v16h aq1 = ld_frag(Qs + arow * 72 + 32 + koffA, Qs + arow * 72 + 32 + koffA + 16);

  float m[8], l[8];
  v8f oc[4];
#pragma unroll
  for (int i = 0; i < 8; i++) { m[i] = -3.0e38f; l[i] = 0.f; }
#pragma unroll
  for (int nt = 0; nt < 4; nt++)
#pragma unroll
    for (int i = 0; i < 8; i++) oc[nt][i] = 0.f;

  for (int jj = 0; jj < 4; jj++) {
    __syncthreads();   // previous chunk fully consumed
    // ---- load + normalize k (threads 0-127) and v (threads 128-255) ----
    {
      int r = tid & 127;
      const _Float16* src = y + (tokbase + jj * 128 + r) * 3072 +
                            ((tid < 128) ? 1024 : 2048) + h * 64;
      v8h vb[8];
      float ss = 0.f;
#pragma unroll
      for (int i = 0; i < 8; i++) {
        vb[i] = *(const v8h*)(src + i * 8);
#pragma unroll
        for (int j = 0; j < 8; j++) { float f = (float)vb[i][j]; ss += f * f; }
      }
      float sc = 1.0f / fmaxf(sqrtf(ss), 1e-12f);
      if (tid < 128) {
#pragma unroll
        for (int i = 0; i < 8; i++) {
          v8h sv;
#pragma unroll
          for (int j = 0; j < 8; j++) sv[j] = (_Float16)((float)vb[i][j] * sc);
          *(v8h*)&Ks[r * 72 + i * 8] = sv;
        }
      } else {
#pragma unroll
        for (int i = 0; i < 8; i++)
#pragma unroll
          for (int j = 0; j < 8; j++)
            Vt[(i * 8 + j) * 136 + r] = (_Float16)((float)vb[i][j] * sc);
      }
    }
    __syncthreads();

    // ---- S(16x128) = q @ k^T ----
    v8f s[8];
#pragma unroll
    for (int nt = 0; nt < 8; nt++)
#pragma unroll
      for (int i = 0; i < 8; i++) s[nt][i] = 0.f;
#pragma unroll
    for (int nt = 0; nt < 8; nt++) {
      const _Float16* kp = Ks + (nt * 16 + lnl) * 72 + (hf ? 8 : 0);
      v16h b0 = ld_frag(kp, kp + 16);
      v16h b1 = ld_frag(kp + 32, kp + 48);
      s[nt] = WMMA(aq0, b0, s[nt]);
      s[nt] = WMMA(aq1, b1, s[nt]);
    }

    // ---- online softmax update; stage P (f16) in per-wave LDS ----
#pragma unroll
    for (int i = 0; i < 8; i++) {
      float mx = -3.0e38f;
#pragma unroll
      for (int nt = 0; nt < 8; nt++) mx = fmaxf(mx, s[nt][i]);
#pragma unroll
      for (int off = 1; off < 16; off <<= 1) mx = fmaxf(mx, __shfl_xor(mx, off, 32));
      float mnew = fmaxf(m[i], mx);
      float corr = __expf(m[i] - mnew);
      m[i] = mnew;
      float rs = 0.f;
      const int prow = (wid * 16 + i + hf * 8) * 136;
#pragma unroll
      for (int nt = 0; nt < 8; nt++) {
        float p = __expf(s[nt][i] - mnew);
        rs += p;
        Ps[prow + nt * 16 + lnl] = (_Float16)p;
      }
#pragma unroll
      for (int off = 1; off < 16; off <<= 1) rs += __shfl_xor(rs, off, 32);
      l[i] = l[i] * corr + rs;
#pragma unroll
      for (int nt = 0; nt < 4; nt++) oc[nt][i] *= corr;
    }
    // same-wave LDS RAW on Ps: ensure stores land before fragment reads
    asm volatile("s_wait_dscnt 0" ::: "memory");

    // ---- O(16x64) += P(16x128) @ V(128x64) ----
#pragma unroll
    for (int kk = 0; kk < 4; kk++) {
      const _Float16* pp = Ps + (wid * 16 + lnl) * 136 + kk * 32 + (hf ? 8 : 0);
      v16h pa = ld_frag(pp, pp + 16);
#pragma unroll
      for (int nt = 0; nt < 4; nt++) {
        const _Float16* vp = Vt + (nt * 16 + lnl) * 136 + kk * 32 + (hf ? 8 : 0);
        v16h vb = ld_frag(vp, vp + 16);
        oc[nt] = WMMA(pa, vb, oc[nt]);
      }
    }
  }

  // ---- finalize: divide by l, store o[token][h*64+d] ----
#pragma unroll
  for (int nt = 0; nt < 4; nt++) {
    int col = h * 64 + nt * 16 + lnl;
#pragma unroll
    for (int i = 0; i < 8; i++) {
      size_t row = tokbase + qt * 128 + wid * 16 + i + hf * 8;
      o[row * 1024 + col] = (_Float16)(oc[nt][i] / l[i]);
    }
  }
}

// ---------------------------------------------------------------------------
// Kernel 3: out[8192 x 1024] = f32( o @ wm^T + bm + x )
// ---------------------------------------------------------------------------
__global__ __launch_bounds__(256)
void out_gemm(const _Float16* __restrict__ o, const float* __restrict__ wm,
              const float* __restrict__ bm, const float* __restrict__ x,
              float* __restrict__ out) {
  __shared__ _Float16 Xs[128 * 40];
  __shared__ _Float16 Ws[128 * 40];
  const int n0 = blockIdx.x * 128, m0 = blockIdx.y * 128;
  const int tid = threadIdx.x, wid = tid >> 5, lane = tid & 31;
  const int lnl = lane & 15, hf = lane >> 4;
  const int wm_ = wid >> 2, wn = wid & 3;

  v8f acc[4][2];
#pragma unroll
  for (int a = 0; a < 4; a++)
#pragma unroll
    for (int b = 0; b < 2; b++)
#pragma unroll
      for (int i = 0; i < 8; i++) acc[a][b][i] = 0.f;

  for (int k0 = 0; k0 < 1024; k0 += 32) {
#pragma unroll
    for (int i = 0; i < 2; i++) {           // o tile: already f16, 2 x v8h/thread
      int idx = tid + i * 256;              // 512 8-half chunks: 128 rows x 4
      int row = idx >> 2, c8 = (idx & 3) * 8;
      *(v8h*)&Xs[row * 40 + c8] = *(const v8h*)(o + (size_t)(m0 + row) * 1024 + k0 + c8);
    }
#pragma unroll
    for (int i = 0; i < 4; i++) {           // wm tile: fp32 -> f16
      int idx = tid + i * 256;
      int row = idx >> 3, c4 = (idx & 7) * 4;
      float4 g = *(const float4*)(wm + (size_t)(n0 + row) * 1024 + k0 + c4);
      v4h h2 = {(_Float16)g.x, (_Float16)g.y, (_Float16)g.z, (_Float16)g.w};
      *(v4h*)&Ws[row * 40 + c4] = h2;
    }
    __syncthreads();
    v16h af[4], bf[2];
#pragma unroll
    for (int mt = 0; mt < 4; mt++) {
      const _Float16* p = Xs + (wm_ * 64 + mt * 16 + lnl) * 40 + (hf ? 8 : 0);
      af[mt] = ld_frag(p, p + 16);
    }
#pragma unroll
    for (int nt = 0; nt < 2; nt++) {
      const _Float16* p = Ws + (wn * 32 + nt * 16 + lnl) * 40 + (hf ? 8 : 0);
      bf[nt] = ld_frag(p, p + 16);
    }
#pragma unroll
    for (int mt = 0; mt < 4; mt++)
#pragma unroll
      for (int nt = 0; nt < 2; nt++)
        acc[mt][nt] = WMMA(af[mt], bf[nt], acc[mt][nt]);
    __syncthreads();
  }

#pragma unroll
  for (int mt = 0; mt < 4; mt++) {
    int rbase = m0 + wm_ * 64 + mt * 16 + (hf ? 8 : 0);
#pragma unroll
    for (int nt = 0; nt < 2; nt++) {
      int col = n0 + wn * 32 + nt * 16 + lnl;
      float bv = bm[col];
#pragma unroll
      for (int i = 0; i < 8; i++) {
        size_t off = (size_t)(rbase + i) * 1024 + col;
        out[off] = acc[mt][nt][i] + bv + x[off];
      }
    }
  }
}

// ---------------------------------------------------------------------------
extern "C" void kernel_launch(void* const* d_in, const int* in_sizes, int n_in,
                              void* d_out, int out_size, void* d_ws, size_t ws_size,
                              hipStream_t stream) {
  const float* x   = (const float*)d_in[0];
  const float* wq  = (const float*)d_in[1];
  const float* bq  = (const float*)d_in[2];
  const float* wkv = (const float*)d_in[3];
  const float* bkv = (const float*)d_in[4];
  const float* wm  = (const float*)d_in[5];
  const float* bm  = (const float*)d_in[6];
  float* out = (float*)d_out;

  _Float16* y = (_Float16*)d_ws;                  // 8192 x 3072 f16 (q|k|v), 48 MB
  _Float16* o = y + (size_t)8192 * 3072;          // 8192 x 1024 f16, 16 MB

  qkv_gemm<<<dim3(24, 64), 256, 0, stream>>>(x, wq, bq, wkv, bkv, y);
  attn<<<dim3(1024), 256, 0, stream>>>(y, o);
  out_gemm<<<dim3(8, 64), 256, 0, stream>>>(o, wm, bm, x, out);
}
